// TuckERT_31086973289235
// MI455X (gfx1250) — matrix operand order repaired and checked by reading, not compiled, not run on hardware
//
#include <hip/hip_runtime.h>
#include <hip/hip_bf16.h>

#define BATCH 512
#define NE_T  100000
#define DE_   128
#define DR_   128
#define DT_   32
#define KQ    (DR_ * DT_)   /* 4096 */
#define NM    (DE_ * DE_)   /* 16384 */
#define LDA   40            /* padded LDS row stride (elems): 20 dwords -> conflict-free */
#define WSTRIDE ((size_t)(128 * 4096))   /* W elems per p-step */

typedef __attribute__((ext_vector_type(16))) __bf16        v16bf;
typedef __attribute__((ext_vector_type(8)))  float         v8f;
typedef __attribute__((ext_vector_type(4)))  float         f32x4;
typedef __attribute__((ext_vector_type(4)))  unsigned int  u32x4;

union ABfrag { v16bf v; u32x4 q[2]; };

// fp32 -> bf16 round-to-nearest-even, bit-level
__device__ __forceinline__ unsigned short f2bf(float f) {
    unsigned int u = __builtin_bit_cast(unsigned int, f);
    u += 0x7FFFu + ((u >> 16) & 1u);
    return (unsigned short)(u >> 16);
}
__device__ __forceinline__ unsigned int packbf(float a, float b) {
    return (unsigned int)f2bf(a) | ((unsigned int)f2bf(b) << 16);
}
// low 32 bits of a generic pointer to __shared__ = LDS offset
__device__ __forceinline__ unsigned lds_off32(const void* p) {
    return (unsigned)(size_t)p;
}

// ---------------------------------------------------------------------------
// Kernel 1: q[b, p*32+u] = R[r_idx[b], p] * T[t_idx[b], u]   (bf16, [512,4096])
// ---------------------------------------------------------------------------
__global__ void build_q(const int* __restrict__ r_idx, const int* __restrict__ t_idx,
                        const float* __restrict__ R, const float* __restrict__ T,
                        unsigned short* __restrict__ q) {
    int idx = blockIdx.x * 256 + threadIdx.x;       // 0 .. 512*4096-1
    int b = idx >> 12;
    int k = idx & 4095;
    int p = k >> 5;
    int u = k & 31;
    float rv = R[(size_t)r_idx[b] * DR_ + p];
    float tv = T[(size_t)t_idx[b] * DT_ + u];
    q[idx] = f2bf(rv * tv);
}

// ---------------------------------------------------------------------------
// Kernel 2: M[b, i*128+j] = sum_{p,u} q[b,p*32+u] * W[p,i,j,u]
// GEMM M=512 K=4096 N=16384. One WG per i-block: 1024 thr = 32 waves, each
// wave owns a 16(M)x128(N) strip -> W streamed from HBM exactly once.
// Distance-2 pipeline: W regs + async A copies issued 2 tiles ahead, so the
// s_wait for tile p+1's data is covered by a full iteration of WMMA work.
// ---------------------------------------------------------------------------
__global__ void __launch_bounds__(1024)
tucker_gemm(const unsigned short* __restrict__ q,
            const float* __restrict__ W,
            float* __restrict__ Mout) {
    __shared__ __align__(16) unsigned short ldsA[3][512 * LDA];  // 3 x 40 KB
    __shared__ __align__(16) unsigned short ldsB[2][128 * LDA];  // 2 x 10 KB

    const int tid  = threadIdx.x;
    const int iblk = blockIdx.x;          // 0..127 (i index -> N block of 128)
    const int wave = tid >> 5;            // 0..31 -> batch rows wave*16..+15
    const int lane = tid & 31;
    const int half = lane >> 4;
    const int l16  = lane & 15;

    v8f c[8];
    v8f zero = {0.f, 0.f, 0.f, 0.f, 0.f, 0.f, 0.f, 0.f};
#pragma unroll
    for (int nt = 0; nt < 8; ++nt) c[nt] = zero;

    const int arow = tid >> 1, apart = tid & 1;  // 512 rows x 2 halves
    const unsigned adst0     = lds_off32(&ldsA[0][arow * LDA + apart * 16]);
    const unsigned aslotB    = (unsigned)(512 * LDA * 2);      // bytes per A slot
    const unsigned short* qrow = q + (size_t)arow * KQ + apart * 16;

    const int e  = tid * 4;               // linear src index = j*32+u = n*32+k
    const int bn = e >> 5, bk = e & 31;
    const int bdst = bn * LDA + bk;
    const float* wcol = W + (size_t)iblk * 4096 + e;

    // ---- prologue: A copies for tiles 0,1; B tile 0 staged, tile 1 in regs ----
    f32x4 wregA;
    {
        asm volatile("global_load_async_to_lds_b128 %0, %1, off\n\t"
                     "global_load_async_to_lds_b128 %0, %1, off offset:16"
                     :: "v"(adst0), "v"(qrow) : "memory");
        asm volatile("global_load_async_to_lds_b128 %0, %1, off\n\t"
                     "global_load_async_to_lds_b128 %0, %1, off offset:16"
                     :: "v"(adst0 + aslotB), "v"(qrow + 32) : "memory");
        f32x4 w0 = *(const f32x4*)(wcol);
        wregA    = *(const f32x4*)(wcol + WSTRIDE);
        ((unsigned int*)ldsB[0])[(bdst >> 1)]     = packbf(w0[0], w0[1]);
        ((unsigned int*)ldsB[0])[(bdst >> 1) + 1] = packbf(w0[2], w0[3]);
    }

    for (int p = 0; p < 128; ++p) {
        const int buf = p & 1, nxt = buf ^ 1;
        // async copies retire in order (2 per tile): <=2 outstanding => tile p done
        if (p < 127) asm volatile("s_wait_asynccnt 0x2" ::: "memory");
        else         asm volatile("s_wait_asynccnt 0x0" ::: "memory");
        __syncthreads();

        // ---- kick off tile p+2: W regs + async A copy (distance 2) ----
        f32x4 wregB = wregA;
        const bool more2 = (p + 2) < 128;
        if (more2) {
            wregB = *(const f32x4*)(wcol + (size_t)(p + 2) * WSTRIDE);
            const unsigned slot = (unsigned)((p + 2) % 3);
            asm volatile("global_load_async_to_lds_b128 %0, %1, off\n\t"
                         "global_load_async_to_lds_b128 %0, %1, off offset:16"
                         :: "v"(adst0 + slot * aslotB), "v"(qrow + (p + 2) * 32)
                         : "memory");
        }

        // ---- compute tile p ----
        ABfrag fa;
        const unsigned short* abase = &ldsA[p % 3][(wave * 16 + l16) * LDA];
        fa.q[0] = *(const u32x4*)(abase + half * 8);
        fa.q[1] = *(const u32x4*)(abase + 16 + half * 8);
#pragma unroll
        for (int nt = 0; nt < 8; ++nt) {
            ABfrag fb;
            const unsigned short* bbase = &ldsB[buf][(nt * 16 + l16) * LDA + half * 8];
            fb.q[0] = *(const u32x4*)(bbase);
            fb.q[1] = *(const u32x4*)(bbase + 16);
            c[nt] = __builtin_amdgcn_wmma_f32_16x16x32_bf16(
                false, fa.v, false, fb.v, (short)0, c[nt], false, false);
        }

        // ---- convert + store tile p+1 (regs loaded a full iteration ago) ----
        if (p + 1 < 128) {
            ((unsigned int*)ldsB[nxt])[(bdst >> 1)]     = packbf(wregA[0], wregA[1]);
            ((unsigned int*)ldsB[nxt])[(bdst >> 1) + 1] = packbf(wregA[2], wregA[3]);
        }
        wregA = wregB;
    }

    // ---- store: lane = N col, VGPR v = row (v + 8*half) ----
#pragma unroll
    for (int nt = 0; nt < 8; ++nt) {
        int n = iblk * 128 + nt * 16 + l16;
#pragma unroll
        for (int v = 0; v < 8; ++v) {
            int m = wave * 16 + half * 8 + v;
            Mout[(size_t)m * NM + n] = c[nt][v];
        }
    }
}

// ---------------------------------------------------------------------------
// Kernel 3: x[b,j] = sum_i E[e1_idx[b], i] * M[b, i*128+j]
// ---------------------------------------------------------------------------
__global__ void reduce_x(const float* __restrict__ Mbuf, const float* __restrict__ E,
                         const int* __restrict__ e1_idx, float* __restrict__ x) {
    __shared__ float e1s[DE_];
    const int b = blockIdx.x;
    const int j = threadIdx.x;  // 128 threads
    e1s[j] = E[(size_t)e1_idx[b] * DE_ + j];
    __syncthreads();
    float acc = 0.f;
    const float* Mrow = Mbuf + (size_t)b * NM;
#pragma unroll 8
    for (int i = 0; i < DE_; ++i) acc += e1s[i] * Mrow[i * DE_ + j];
    x[b * DE_ + j] = acc;
}

// ---------------------------------------------------------------------------
// Kernel 4: out[b,n] = sigmoid( sum_k x[b,k] * E[n,k] )
// GEMM M=512 K=128 N=100000. One WG per n-block (782), 32 waves cover all 512
// rows -> E streamed exactly once. Distance-2 pipeline on the E stream.
// ---------------------------------------------------------------------------
__global__ void __launch_bounds__(1024)
logits_gemm(const float* __restrict__ x, const float* __restrict__ E,
            float* __restrict__ out) {
    __shared__ __align__(16) unsigned short ldsA[2][512 * LDA];  // 2 x 40 KB
    __shared__ __align__(16) unsigned short ldsB[2][128 * LDA];  // 2 x 10 KB

    const int tid  = threadIdx.x;
    const int nblk = blockIdx.x;          // 0..781
    const int wave = tid >> 5;
    const int lane = tid & 31;
    const int half = lane >> 4;
    const int l16  = lane & 15;
    const int n0   = nblk * 128;

    v8f c[8];
    v8f zero = {0.f, 0.f, 0.f, 0.f, 0.f, 0.f, 0.f, 0.f};
#pragma unroll
    for (int nt = 0; nt < 8; ++nt) c[nt] = zero;

    const int row = tid >> 1, part = tid & 1;       // A: 512 rows x 2 halves
    const int e  = tid * 4;                         // B: n = e>>5, k = e&31
    const int bn = e >> 5, bk = e & 31;
    const int bdst = bn * LDA + bk;
    const bool ok = (n0 + bn) < NE_T;
    const float* bsrc = E + (size_t)(n0 + bn) * DE_ + bk;
    const f32x4 z4 = {0.f, 0.f, 0.f, 0.f};

    // ---- prologue: A+B for ks=0 staged; B for ks=1 in regs ----
    f32x4 bregA;
    {
#pragma unroll
        for (int f = 0; f < 4; ++f) {
            f32x4 v = *(const f32x4*)(x + (size_t)row * DE_ + part * 16 + f * 4);
            int d = row * LDA + part * 16 + f * 4;
            ((unsigned int*)ldsA[0])[(d >> 1)]     = packbf(v[0], v[1]);
            ((unsigned int*)ldsA[0])[(d >> 1) + 1] = packbf(v[2], v[3]);
        }
        f32x4 b0 = ok ? *(const f32x4*)(bsrc) : z4;
        bregA    = ok ? *(const f32x4*)(bsrc + 32) : z4;
        ((unsigned int*)ldsB[0])[(bdst >> 1)]     = packbf(b0[0], b0[1]);
        ((unsigned int*)ldsB[0])[(bdst >> 1) + 1] = packbf(b0[2], b0[3]);
    }

    for (int ks = 0; ks < 4; ++ks) {
        const int buf = ks & 1, nxt = buf ^ 1;
        __syncthreads();

        // ---- prefetch: A regs for ks+1 (L2-hot), E regs for ks+2 ----
        f32x4 an[4], bregB = bregA;
        const bool more  = (ks + 1) < 4;
        const bool more2 = (ks + 2) < 4;
        if (more) {
#pragma unroll
            for (int f = 0; f < 4; ++f)
                an[f] = *(const f32x4*)(x + (size_t)row * DE_ + (ks + 1) * 32 + part * 16 + f * 4);
        }
        if (more2)
            bregB = ok ? *(const f32x4*)(bsrc + (ks + 2) * 32) : z4;

        // ---- compute ks ----
        ABfrag fa;
        const unsigned short* abase = &ldsA[buf][(wave * 16 + l16) * LDA];
        fa.q[0] = *(const u32x4*)(abase + half * 8);
        fa.q[1] = *(const u32x4*)(abase + 16 + half * 8);
#pragma unroll
        for (int nt = 0; nt < 8; ++nt) {
            ABfrag fb;
            const unsigned short* bbase = &ldsB[buf][(nt * 16 + l16) * LDA + half * 8];
            fb.q[0] = *(const u32x4*)(bbase);
            fb.q[1] = *(const u32x4*)(bbase + 16);
            c[nt] = __builtin_amdgcn_wmma_f32_16x16x32_bf16(
                false, fa.v, false, fb.v, (short)0, c[nt], false, false);
        }

        // ---- convert + store ks+1 into other buffer ----
        if (more) {
#pragma unroll
            for (int f = 0; f < 4; ++f) {
                int d = row * LDA + part * 16 + f * 4;
                ((unsigned int*)ldsA[nxt])[(d >> 1)]     = packbf(an[f][0], an[f][1]);
                ((unsigned int*)ldsA[nxt])[(d >> 1) + 1] = packbf(an[f][2], an[f][3]);
            }
            ((unsigned int*)ldsB[nxt])[(bdst >> 1)]     = packbf(bregA[0], bregA[1]);
            ((unsigned int*)ldsB[nxt])[(bdst >> 1) + 1] = packbf(bregA[2], bregA[3]);
        }
        bregA = bregB;
    }

    // ---- sigmoid + guarded store ----
#pragma unroll
    for (int nt = 0; nt < 8; ++nt) {
        int n = n0 + nt * 16 + l16;
        if (n < NE_T) {
#pragma unroll
            for (int v = 0; v < 8; ++v) {
                int m = wave * 16 + half * 8 + v;
                out[(size_t)m * NE_T + n] = 1.f / (1.f + __expf(-c[nt][v]));
            }
        }
    }
}

// ---------------------------------------------------------------------------
extern "C" void kernel_launch(void* const* d_in, const int* in_sizes, int n_in,
                              void* d_out, int out_size, void* d_ws, size_t ws_size,
                              hipStream_t stream) {
    (void)in_sizes; (void)n_in; (void)out_size; (void)ws_size;

    const int*   e1_idx = (const int*)d_in[0];
    const int*   r_idx  = (const int*)d_in[1];
    const int*   t_idx  = (const int*)d_in[2];
    const float* E      = (const float*)d_in[3];   // [100000,128]
    const float* R      = (const float*)d_in[4];   // [500,128]
    const float* T      = (const float*)d_in[5];   // [365,32]
    const float* W      = (const float*)d_in[6];   // [128,128,128,32]
    float* out = (float*)d_out;                    // [512,100000]

    char* ws = (char*)d_ws;
    unsigned short* q  = (unsigned short*)ws;                    // 512*4096 bf16 = 4 MB
    float* Mbuf        = (float*)(ws + (4u << 20));              // 512*16384 f32 = 32 MB
    float* x           = (float*)(ws + (36u << 20));             // 512*128  f32 = 256 KB

    build_q<<<(BATCH * KQ) / 256, 256, 0, stream>>>(r_idx, t_idx, R, T, q);
    tucker_gemm<<<128, 1024, 0, stream>>>(q, W, Mbuf);
    reduce_x<<<BATCH, DE_, 0, stream>>>(Mbuf, E, e1_idx, x);
    logits_gemm<<<782, 1024, 0, stream>>>(x, E, out);
}
